// DinoDecoderBlock_74500502716814
// MI455X (gfx1250) — matrix operand order repaired and checked by reading, not compiled, not run on hardware
//
#include <hip/hip_runtime.h>
#include <hip/hip_bf16.h>
#include <cstdint>
#include <cstddef>

// ---------------------------------------------------------------------------
// DINO decoder block for MI455X (gfx1250, wave32, WMMA bf16 path)
//   - all GEMMs / attention einsums on v_wmma_f32_16x16x32_bf16 (f32 accum)
//   - GEMM A-tiles staged to LDS via global_load_async_to_lds_b128 (ASYNCcnt),
//     double-buffered per wave; B fragments are contiguous 32B global loads
//   - flash-style attention with online softmax, P reshaped via per-wave LDS
// ---------------------------------------------------------------------------

typedef __attribute__((ext_vector_type(16))) __bf16 v16bf;
typedef __attribute__((ext_vector_type(8)))  __bf16 v8bf;
typedef __attribute__((ext_vector_type(8)))  float  v8f;
typedef unsigned short u16t;
typedef unsigned int   u32t;

#define BQ      8
#define NQS     1024
#define NKS     1024
#define CDIM    768
#define NHEADS  12
#define HDIM    64
#define HIDDEN  3072
#define MROWS   (BQ * NQS)   /* 8192 */

enum { GF_BIAS = 1, GF_ADD = 2, GF_GELU = 4, GF_OUTBF16 = 8 };

// ---------------- small helpers ----------------

__device__ __forceinline__ u16t f2bf(float f) {
  u32t u = __float_as_uint(f);
  u32t r = u + 0x7FFFu + ((u >> 16) & 1u);   // round-to-nearest-even
  return (u16t)(r >> 16);
}
__device__ __forceinline__ float bf2f(u16t h) {
  return __uint_as_float(((u32t)h) << 16);
}
__device__ __forceinline__ v8bf ld8(const u16t* p) {
  return *reinterpret_cast<const v8bf*>(p);
}
__device__ __forceinline__ v16bf ld16(const u16t* p) {
  return *reinterpret_cast<const v16bf*>(p);
}
__device__ __forceinline__ v16bf cat16(v8bf a, v8bf b) {
  v16bf r;
#pragma unroll
  for (int i = 0; i < 8; ++i) { r[i] = a[i]; r[i + 8] = b[i]; }
  return r;
}
__device__ __forceinline__ v8f vzero8() {
  v8f z;
#pragma unroll
  for (int i = 0; i < 8; ++i) z[i] = 0.f;
  return z;
}
__device__ __forceinline__ v8f wmma_bf16(v16bf a, v16bf b, v8f c) {
  // D = A(16x32) * B(32x16) + C, f32 accumulation
  return __builtin_amdgcn_wmma_f32_16x16x32_bf16(false, a, false, b, (short)0, c,
                                                 false, false);
}

// A fragment (16xK tile rows mbase..mbase+15, K-chunk k0..k0+31), row stride ldk.
// Lane layout (contiguous, matching frag_b): lanes 0-15 hold K[0..15];
// lanes 16-31 hold K[16..31].
__device__ __forceinline__ v16bf frag_a(const u16t* A, int ldk, int mbase, int k0,
                                        int lane) {
  const u16t* p = A + (size_t)(mbase + (lane & 15)) * ldk + k0 + ((lane >> 4) << 4);
  return cat16(ld8(p), ld8(p + 8));
}
// B fragment from row-major W[N][K] representing B = W^T (B[k][n] = W[n][k]).
// Lane L: column n = L&15, koff = (L>>4)*16 -> one contiguous 32-byte load.
__device__ __forceinline__ v16bf frag_b(const u16t* W, int ldk, int nbase, int k0,
                                        int lane) {
  const u16t* p = W + (size_t)(nbase + (lane & 15)) * ldk + k0 + ((lane >> 4) << 4);
  return ld16(p);
}

// ---------------- elementwise f32 -> bf16 ----------------

__global__ void f32_to_bf16_kernel(const float* __restrict__ in,
                                   u16t* __restrict__ out, int n) {
  int i = blockIdx.x * 256 + threadIdx.x;
  if (i < n) out[i] = f2bf(in[i]);
}

// ---------------- LayerNorm (f32 in, bf16 out), one wave per row ----------------

__global__ __launch_bounds__(256) void ln_bf16_kernel(
    const float* __restrict__ x, const float* __restrict__ g,
    const float* __restrict__ b, u16t* __restrict__ out) {
  const int lane = threadIdx.x & 31;
  const int row  = blockIdx.x * 8 + (threadIdx.x >> 5);
  const float* p = x + (size_t)row * CDIM;
  float v[24];
  float s = 0.f;
#pragma unroll
  for (int i = 0; i < 24; ++i) { v[i] = p[lane + 32 * i]; s += v[i]; }
#pragma unroll
  for (int o = 16; o > 0; o >>= 1) s += __shfl_xor(s, o, 32);
  const float mean = s * (1.f / (float)CDIM);
  float q = 0.f;
#pragma unroll
  for (int i = 0; i < 24; ++i) { float d = v[i] - mean; q += d * d; }
#pragma unroll
  for (int o = 16; o > 0; o >>= 1) q += __shfl_xor(q, o, 32);
  const float inv = rsqrtf(q * (1.f / (float)CDIM) + 1e-5f);
  u16t* op = out + (size_t)row * CDIM;
#pragma unroll
  for (int i = 0; i < 24; ++i) {
    const int c = lane + 32 * i;
    op[c] = f2bf((v[i] - mean) * inv * g[c] + b[c]);
  }
}

// ---------------- GEMM: Out[M,N] = A[M,K](bf16) @ W[N,K]^T (+bias,+gelu,+res) ----
// Wave tile 32(M) x 64(N). Each wave double-buffers its own 32x32 A-tile in LDS
// using global_load_async_to_lds_b128 (4 issues/stage, 16B/lane, coalesced rows).

__global__ __launch_bounds__(256) void gemm_bf16_kernel(
    const u16t* __restrict__ A, const u16t* __restrict__ W,
    const float* __restrict__ bias, const float* __restrict__ addsrc,
    void* __restrict__ out, int M, int N, int K, int flags) {
  __shared__ u16t ldsA[8][2][32 * 32];   // 8 waves x 2 buffers x 2KB = 32KB

  const int wave = threadIdx.x >> 5;
  const int lane = threadIdx.x & 31;
  const int n0 = blockIdx.x * 64;
  const int m0 = blockIdx.y * 256 + wave * 32;

  // async staging addressing: instruction q copies rows m0+q*8 .. +q*8+7;
  // lane -> (row = q*8 + lane/4, 16B column chunk = (lane&3)*16B)
  const int arow = lane >> 2;               // 0..7
  const int acol = (lane & 3) * 8;          // element offset (16B)
  const u16t* gA = A + (size_t)(m0 + arow) * K + acol;
  const u32t lb0 = (u32t)(uintptr_t)&ldsA[wave][0][0];
  const u32t lb1 = (u32t)(uintptr_t)&ldsA[wave][1][0];
  const u32t loff = (u32t)(arow * 32 + acol) * 2;   // byte offset inside buffer

  v8f acc[2][4];
#pragma unroll
  for (int i = 0; i < 2; ++i)
#pragma unroll
    for (int j = 0; j < 4; ++j) acc[i][j] = vzero8();

  // prologue: stage k=0 into buffer 0
  {
#pragma unroll
    for (int q = 0; q < 4; ++q) {
      const u16t* gp = gA + (size_t)(q * 8) * K;
      u32t lp = lb0 + loff + (u32t)(q * 8 * 64);
      asm volatile("global_load_async_to_lds_b128 %0, %1, off"
                   :: "v"(lp), "v"(gp) : "memory");
    }
  }

  int cur = 0;
  for (int k0 = 0; k0 < K; k0 += 32) {
    const bool more = (k0 + 32) < K;
    if (more) {
      const u32t lb = cur ? lb0 : lb1;   // stage into the other buffer
#pragma unroll
      for (int q = 0; q < 4; ++q) {
        const u16t* gp = gA + (size_t)(q * 8) * K + (k0 + 32);
        u32t lp = lb + loff + (u32t)(q * 8 * 64);
        asm volatile("global_load_async_to_lds_b128 %0, %1, off"
                     :: "v"(lp), "v"(gp) : "memory");
      }
      // previous 4 issues (current buffer) complete when count drops to <=4
      asm volatile("s_wait_asynccnt 0x4" ::: "memory");
    } else {
      asm volatile("s_wait_asynccnt 0x0" ::: "memory");
    }
    const u16t* lA = cur ? &ldsA[wave][1][0] : &ldsA[wave][0][0];
    v16bf a0 = frag_a(lA, 32, 0,  0, lane);
    v16bf a1 = frag_a(lA, 32, 16, 0, lane);
    // fence ds reads so next iteration may safely overwrite this buffer
    asm volatile("s_wait_dscnt 0x0" ::: "memory");
#pragma unroll
    for (int j = 0; j < 4; ++j) {
      v16bf b = frag_b(W, K, n0 + 16 * j, k0, lane);
      acc[0][j] = wmma_bf16(a0, b, acc[0][j]);
      acc[1][j] = wmma_bf16(a1, b, acc[1][j]);
    }
    cur ^= 1;
  }

  const int nn = lane & 15;
  const int hf = lane >> 4;
#pragma unroll
  for (int i = 0; i < 2; ++i)
#pragma unroll
    for (int j = 0; j < 4; ++j) {
      const int n = n0 + 16 * j + nn;
      const float bv = (flags & GF_BIAS) ? bias[n] : 0.f;
#pragma unroll
      for (int r = 0; r < 8; ++r) {
        const int m = m0 + 16 * i + 8 * hf + r;   // C-layout: lanes16-31 => M+8
        float v = acc[i][j][r] + bv;
        if (flags & GF_GELU) v = 0.5f * v * (1.f + erff(v * 0.70710678118654752f));
        if (flags & GF_ADD) v += addsrc[(size_t)m * N + n];
        if (flags & GF_OUTBF16) ((u16t*)out)[(size_t)m * N + n] = f2bf(v);
        else                    ((float*)out)[(size_t)m * N + n] = v;
      }
    }
}

// ---------------- split heads: proj rows -> [B,H,N,D] (q,k) and [B,H,D,N] (v) ----

__global__ __launch_bounds__(256) void split_heads_kernel(
    const u16t* __restrict__ qsrc, const u16t* __restrict__ ksrc,
    const u16t* __restrict__ vsrc, int srcld, u16t* __restrict__ qh,
    u16t* __restrict__ kh, u16t* __restrict__ vth) {
  int idx = blockIdx.x * 256 + threadIdx.x;
  const int total = BQ * NHEADS * NQS * HDIM;
  if (idx >= total) return;
  const int d = idx & 63;
  int t = idx >> 6;
  const int n = t & (NQS - 1); t >>= 10;
  const int h = t % NHEADS;
  const int bb = t / NHEADS;
  const size_t s = (size_t)(bb * NQS + n) * srcld + h * HDIM + d;
  const size_t bh = (size_t)(bb * NHEADS + h);
  // fold SCALE = HD^-0.5 = 0.125 into q
  qh[(bh * NQS + n) * HDIM + d] = f2bf(bf2f(qsrc[s]) * 0.125f);
  kh[(bh * NQS + n) * HDIM + d] = ksrc[s];
  vth[(bh * HDIM + d) * NKS + n] = vsrc[s];   // V transposed for contiguous B-frags
}

// ---------------- flash attention (one wave = 16 q rows, kv tiles of 64) -------

__global__ __launch_bounds__(256) void flash_attn_kernel(
    const u16t* __restrict__ qh, const u16t* __restrict__ kh,
    const u16t* __restrict__ vth, const unsigned char* __restrict__ mask,
    u16t* __restrict__ o) {
  __shared__ u16t plds[8][16 * 80];   // per-wave P tile, stride 80 (16B aligned)
  const int wave = threadIdx.x >> 5;
  const int lane = threadIdx.x & 31;
  const int nn = lane & 15;
  const int hf = lane >> 4;
  const int bh = blockIdx.y;
  const int bb = bh / NHEADS;
  const int hh = bh % NHEADS;
  const int qbase = blockIdx.x * 128 + wave * 16;
  const u16t* Q  = qh  + (size_t)bh * NQS * HDIM;
  const u16t* Km = kh  + (size_t)bh * NKS * HDIM;
  const u16t* Vt = vth + (size_t)bh * HDIM * NKS;
  u16t* pl = plds[wave];

  // Q fragments (M16 x K64 -> two 16x32 A-frags), resident for whole kv loop
  const v16bf qa0 = frag_a(Q, HDIM, qbase, 0,  lane);
  const v16bf qa1 = frag_a(Q, HDIM, qbase, 32, lane);

  v8f oacc[4];
#pragma unroll
  for (int j = 0; j < 4; ++j) oacc[j] = vzero8();
  float rmax[8], rsum[8];
#pragma unroll
  for (int r = 0; r < 8; ++r) { rmax[r] = -3.0e38f; rsum[r] = 0.f; }

  for (int kv = 0; kv < NKS; kv += 64) {
    // S = Q * K^T  (16 x 64 tile)
    v8f s[4];
#pragma unroll
    for (int j = 0; j < 4; ++j) {
      v8f sj = vzero8();
      sj = wmma_bf16(qa0, frag_b(Km, HDIM, kv + 16 * j, 0,  lane), sj);
      sj = wmma_bf16(qa1, frag_b(Km, HDIM, kv + 16 * j, 32, lane), sj);
      s[j] = sj;
    }
    if (mask) {
#pragma unroll
      for (int j = 0; j < 4; ++j) {
        const int n = kv + 16 * j + nn;
#pragma unroll
        for (int r = 0; r < 8; ++r) {
          const int m = qbase + 8 * hf + r;
          if (!mask[(size_t)m * NKS + n]) s[j][r] = -3.0e38f;
        }
      }
    }
    // online softmax: each row lives on 16 lanes (half-wave), 4 values/lane
#pragma unroll
    for (int r = 0; r < 8; ++r) {
      float mx = fmaxf(fmaxf(s[0][r], s[1][r]), fmaxf(s[2][r], s[3][r]));
#pragma unroll
      for (int off = 8; off > 0; off >>= 1) mx = fmaxf(mx, __shfl_xor(mx, off, 32));
      const float mnew = fmaxf(rmax[r], mx);
      const float corr = __expf(rmax[r] - mnew);
      rmax[r] = mnew;
      float ps = 0.f;
#pragma unroll
      for (int j = 0; j < 4; ++j) {
        const float pv = __expf(s[j][r] - mnew);
        s[j][r] = pv;
        ps += pv;
      }
#pragma unroll
      for (int off = 8; off > 0; off >>= 1) ps += __shfl_xor(ps, off, 32);
      rsum[r] = rsum[r] * corr + ps;
#pragma unroll
      for (int j = 0; j < 4; ++j) oacc[j][r] *= corr;
    }
    // C-layout -> A-layout reshape of P through this wave's LDS tile
#pragma unroll
    for (int j = 0; j < 4; ++j)
#pragma unroll
      for (int r = 0; r < 8; ++r)
        pl[(8 * hf + r) * 80 + 16 * j + nn] = f2bf(s[j][r]);
    asm volatile("s_wait_dscnt 0x0" ::: "memory");
    const v16bf pa0 = frag_a(pl, 80, 0, 0,  lane);
    const v16bf pa1 = frag_a(pl, 80, 0, 32, lane);
    // O += P * V   (V transposed => contiguous B-frags)
#pragma unroll
    for (int j = 0; j < 4; ++j) {
      oacc[j] = wmma_bf16(pa0, frag_b(Vt, NKS, 16 * j, kv,      lane), oacc[j]);
      oacc[j] = wmma_bf16(pa1, frag_b(Vt, NKS, 16 * j, kv + 32, lane), oacc[j]);
    }
  }
  // normalize + write merged-head layout [B*NQ, C]
#pragma unroll
  for (int r = 0; r < 8; ++r) {
    const float inv = 1.f / rsum[r];
    const int m = qbase + 8 * hf + r;
    u16t* op = o + (size_t)(bb * NQS + m) * CDIM + hh * HDIM;
#pragma unroll
    for (int j = 0; j < 4; ++j) op[16 * j + nn] = f2bf(oacc[j][r] * inv);
  }
}

// ---------------------------------------------------------------------------
// host orchestration
// ---------------------------------------------------------------------------

extern "C" void kernel_launch(void* const* d_in, const int* in_sizes, int n_in,
                              void* d_out, int out_size, void* d_ws, size_t ws_size,
                              hipStream_t stream) {
  (void)in_sizes; (void)n_in; (void)out_size; (void)ws_size;
  const float* x           = (const float*)d_in[0];
  const float* y           = (const float*)d_in[1];
  const unsigned char* msk = (const unsigned char*)d_in[4];
  const float* qkv_w       = (const float*)d_in[5];
  const float* attn_proj_w = (const float*)d_in[6];
  const float* attn_proj_b = (const float*)d_in[7];
  const float* q_w         = (const float*)d_in[8];
  const float* k_w         = (const float*)d_in[9];
  const float* v_w         = (const float*)d_in[10];
  const float* ca_proj_w   = (const float*)d_in[11];
  const float* ca_proj_b   = (const float*)d_in[12];
  const float* fc1_w       = (const float*)d_in[13];
  const float* fc1_b       = (const float*)d_in[14];
  const float* fc2_w       = (const float*)d_in[15];
  const float* fc2_b       = (const float*)d_in[16];
  const float* ln1_g       = (const float*)d_in[17];
  const float* ln1_b       = (const float*)d_in[18];
  const float* ln2_g       = (const float*)d_in[19];
  const float* ln2_b       = (const float*)d_in[20];
  const float* ln3_g       = (const float*)d_in[21];
  const float* ln3_b       = (const float*)d_in[22];
  const float* lny_g       = (const float*)d_in[23];
  const float* lny_b       = (const float*)d_in[24];
  float* out = (float*)d_out;

  char* ws = (char*)d_ws;
  size_t off = 0;
  auto take = [&](size_t bytes) -> char* {
    char* p = ws + off;
    off = (off + bytes + 255) & ~(size_t)255;
    return p;
  };
  const size_t MC = (size_t)MROWS * CDIM;   // 6,291,456 elements
  u16t* Wqkv = (u16t*)take((size_t)3 * CDIM * CDIM * 2);
  u16t* Wq   = (u16t*)take((size_t)CDIM * CDIM * 2);
  u16t* Wk   = (u16t*)take((size_t)CDIM * CDIM * 2);
  u16t* Wv   = (u16t*)take((size_t)CDIM * CDIM * 2);
  u16t* Wsp  = (u16t*)take((size_t)CDIM * CDIM * 2);
  u16t* Wcp  = (u16t*)take((size_t)CDIM * CDIM * 2);
  u16t* Wf1  = (u16t*)take((size_t)HIDDEN * CDIM * 2);
  u16t* Wf2  = (u16t*)take((size_t)HIDDEN * CDIM * 2);
  float* xcur = (float*)take(MC * 4);
  u16t* lnA  = (u16t*)take(MC * 2);
  u16t* lnB  = (u16t*)take(MC * 2);
  u16t* qhB  = (u16t*)take(MC * 2);
  u16t* khB  = (u16t*)take(MC * 2);
  u16t* vtB  = (u16t*)take(MC * 2);
  u16t* obuf = (u16t*)take(MC * 2);
  u16t* big0 = (u16t*)take((size_t)MROWS * HIDDEN * 2);

  // y passes through unchanged; start residual stream from x
  hipMemcpyAsync(out + MC, y, MC * 4, hipMemcpyDeviceToDevice, stream);
  hipMemcpyAsync(xcur, x, MC * 4, hipMemcpyDeviceToDevice, stream);

  auto conv = [&](const float* src, u16t* dst, size_t n) {
    f32_to_bf16_kernel<<<dim3((unsigned)((n + 255) / 256)), dim3(256), 0, stream>>>(
        src, dst, (int)n);
  };
  conv(qkv_w, Wqkv, (size_t)3 * CDIM * CDIM);
  conv(q_w, Wq, (size_t)CDIM * CDIM);
  conv(k_w, Wk, (size_t)CDIM * CDIM);
  conv(v_w, Wv, (size_t)CDIM * CDIM);
  conv(attn_proj_w, Wsp, (size_t)CDIM * CDIM);
  conv(ca_proj_w, Wcp, (size_t)CDIM * CDIM);
  conv(fc1_w, Wf1, (size_t)HIDDEN * CDIM);
  conv(fc2_w, Wf2, (size_t)HIDDEN * CDIM);

  const dim3 blk(256);
  const dim3 lngrid(MROWS / 8);
  const dim3 g768(CDIM / 64, MROWS / 256);
  const dim3 g2304((3 * CDIM) / 64, MROWS / 256);
  const dim3 g3072(HIDDEN / 64, MROWS / 256);
  const dim3 sgrid((BQ * NHEADS * NQS * HDIM) / 256);
  const dim3 fgrid(NQS / 128, BQ * NHEADS);

  // ---- self attention ----
  ln_bf16_kernel<<<lngrid, blk, 0, stream>>>(xcur, ln1_g, ln1_b, lnA);
  gemm_bf16_kernel<<<g2304, blk, 0, stream>>>(lnA, Wqkv, nullptr, nullptr, big0,
                                              MROWS, 3 * CDIM, CDIM, GF_OUTBF16);
  split_heads_kernel<<<sgrid, blk, 0, stream>>>(big0, big0 + CDIM, big0 + 2 * CDIM,
                                                3 * CDIM, qhB, khB, vtB);
  flash_attn_kernel<<<fgrid, blk, 0, stream>>>(qhB, khB, vtB, nullptr, obuf);
  gemm_bf16_kernel<<<g768, blk, 0, stream>>>(obuf, Wsp, attn_proj_b, xcur, xcur,
                                             MROWS, CDIM, CDIM, GF_BIAS | GF_ADD);

  // ---- cross attention ----
  ln_bf16_kernel<<<lngrid, blk, 0, stream>>>(xcur, ln2_g, ln2_b, lnA);
  ln_bf16_kernel<<<lngrid, blk, 0, stream>>>(y, lny_g, lny_b, lnB);
  gemm_bf16_kernel<<<g768, blk, 0, stream>>>(lnA, Wq, nullptr, nullptr, big0,
                                             MROWS, CDIM, CDIM, GF_OUTBF16);
  gemm_bf16_kernel<<<g768, blk, 0, stream>>>(lnB, Wk, nullptr, nullptr, big0 + MC,
                                             MROWS, CDIM, CDIM, GF_OUTBF16);
  gemm_bf16_kernel<<<g768, blk, 0, stream>>>(lnB, Wv, nullptr, nullptr,
                                             big0 + 2 * MC, MROWS, CDIM, CDIM,
                                             GF_OUTBF16);
  split_heads_kernel<<<sgrid, blk, 0, stream>>>(big0, big0 + MC, big0 + 2 * MC,
                                                CDIM, qhB, khB, vtB);
  flash_attn_kernel<<<fgrid, blk, 0, stream>>>(qhB, khB, vtB, msk, obuf);
  gemm_bf16_kernel<<<g768, blk, 0, stream>>>(obuf, Wcp, ca_proj_b, xcur, xcur,
                                             MROWS, CDIM, CDIM, GF_BIAS | GF_ADD);

  // ---- MLP ----
  ln_bf16_kernel<<<lngrid, blk, 0, stream>>>(xcur, ln3_g, ln3_b, lnA);
  gemm_bf16_kernel<<<g3072, blk, 0, stream>>>(lnA, Wf1, fc1_b, nullptr, big0, MROWS,
                                              HIDDEN, CDIM,
                                              GF_BIAS | GF_GELU | GF_OUTBF16);
  gemm_bf16_kernel<<<g768, blk, 0, stream>>>(big0, Wf2, fc2_b, xcur, out, MROWS,
                                             CDIM, HIDDEN, GF_BIAS | GF_ADD);
}
